// GraphConv_47373489275409
// MI455X (gfx1250) — matrix operand order repaired and checked by reading, not compiled, not run on hardware
//
#include <hip/hip_runtime.h>

#define BATCH 8
#define NN    2048
#define CC    128
#define NCHUNK (NN / 32)

typedef __bf16        v16bf  __attribute__((ext_vector_type(16)));
typedef __bf16        bf16x2 __attribute__((ext_vector_type(2)));
typedef float         v8f    __attribute__((ext_vector_type(8)));
typedef unsigned int  u32x4  __attribute__((ext_vector_type(4)));
typedef int           i32x4  __attribute__((ext_vector_type(4)));
typedef float         f32x4  __attribute__((ext_vector_type(4)));

union V16U {
    v16bf          v;
    __bf16         h[16];
    u32x4          q[2];
    unsigned int   u[8];
    unsigned short s[16];
};

// two floats -> packed bf16 pair in one v_cvt_pk_bf16_f32
__device__ __forceinline__ unsigned int packbf2(float lo, float hi) {
#if __has_builtin(__builtin_amdgcn_cvt_pk_bf16_f32)
    bf16x2 p = __builtin_amdgcn_cvt_pk_bf16_f32(lo, hi);
    return __builtin_bit_cast(unsigned int, p);
#else
    union { bf16x2 h; unsigned int u; } p;
    p.h[0] = (__bf16)lo;
    p.h[1] = (__bf16)hi;
    return p.u;
#endif
}

// input: h = 0.5*(sj + sk + bias); output: sigmoid(2h) = 0.5*tanh(h) + 0.5
__device__ __forceinline__ float sigmoid_from_half_arg(float h) {
#if __has_builtin(__builtin_amdgcn_tanhf)
    return __builtin_amdgcn_tanhf(h) * 0.5f + 0.5f;        // 1 trans + 1 fma
#else
    float e = __builtin_amdgcn_exp2f(h * -2.8853900817779268f);
    return __builtin_amdgcn_rcpf(1.0f + e);
#endif
}

// AND-mask 0x0000FFFF/0xFFFF0000/0xFFFFFFFF from adjacency bits (0/1), no int mul
__device__ __forceinline__ unsigned int pairmask(int a0, int a1) {
    unsigned int v = (unsigned int)a0 | ((unsigned int)a1 << 16);  // v_lshl_or_b32
    return (v << 16) - v;                                          // == v * 0xFFFF
}
__device__ __forceinline__ unsigned int masked_pair(float s0, float s1, int a0, int a1) {
    return packbf2(s0, s1) & pairmask(a0, a1);                     // exact 0 where adj==0
}

// ---------------------------------------------------------------------------
// Kernel 1: sj[b,n] = x[b,n,:]·w1, sk[b,n] = x[b,n,:]·w2
// ---------------------------------------------------------------------------
__global__ __launch_bounds__(256) void gc_scores(const float* __restrict__ x,
                                                 const float* __restrict__ W_fc,
                                                 float* __restrict__ sj,
                                                 float* __restrict__ sk) {
    int row = blockIdx.x * blockDim.x + threadIdx.x;     // 0 .. B*N-1
    const float* xr = x + (size_t)row * CC;
    float a = 0.f, c = 0.f;
#pragma unroll
    for (int i = 0; i < CC; i += 4) {
        f32x4 xv = *(const f32x4*)&xr[i];
        f32x4 w1 = *(const f32x4*)&W_fc[i];
        f32x4 w2 = *(const f32x4*)&W_fc[CC + i];
        a += xv[0]*w1[0] + xv[1]*w1[1] + xv[2]*w1[2] + xv[3]*w1[3];
        c += xv[0]*w2[0] + xv[1]*w2[1] + xv[2]*w2[2] + xv[3]*w2[3];
    }
    sj[row] = a;
    sk[row] = c;
}

// ---------------------------------------------------------------------------
// Kernel 2: fused  score -> mask -> (w @ x) & rowsum -> normalize -> (@ W_lin)+b
// block = 128 threads (4 waves). wave w owns j rows [j0+16w, j0+16w+16).
// grid = (NN/64, BATCH). Double-buffered x chunk, one barrier per chunk.
// ---------------------------------------------------------------------------
__global__ __launch_bounds__(128) void gc_fused(const float* __restrict__ x,
                                                const int*   __restrict__ adj,
                                                const float* __restrict__ sjg,
                                                const float* __restrict__ skg,
                                                const float* __restrict__ bfc,
                                                const float* __restrict__ Wlin,
                                                const float* __restrict__ blin,
                                                float*       __restrict__ out) {
    __shared__ __attribute__((aligned(32))) unsigned short wl[4][8][32][16];     // 32 KB  W_lin, B layout
    __shared__ __attribute__((aligned(32))) unsigned short xb[2][8][32][16];     // 16 KB  x chunk, double-buffered
    __shared__ __attribute__((aligned(16))) float skl[NN];                       // 8 KB   0.5*(sk+bias)
    __shared__ float sjl[64];
    __shared__ __attribute__((aligned(16))) __bf16 supp[4][16][128];             // 16 KB

    const int b    = blockIdx.y;
    const int j0   = blockIdx.x * 64;
    const int t    = threadIdx.x;        // 0..127
    const int lane = t & 31;
    const int wave = t >> 5;
    const int n    = lane & 15;
    const int hi   = (lane >> 4) & 1;

    const float bias = bfc[0];

    // ---- stage 0.5*(sk[b,:] + bias) ----
#pragma unroll
    for (int r = 0; r < 16; ++r)
        skl[r * 128 + t] = 0.5f * (skg[b * NN + r * 128 + t] + bias);
    if (t < 64) sjl[t] = sjg[b * NN + j0 + t];
    // ---- stage W_lin into B layout ----
#pragma unroll
    for (int g = 0; g < 16; ++g) {
        int G  = g * 128 + t;            // 0..2047 pair-groups
        int k  = (G >> 5) * 2;
        int c0 = (G & 31) * 4;
        f32x4 wlo = *(const f32x4*)&Wlin[k * CC + c0];
        f32x4 whi = *(const f32x4*)&Wlin[(k + 1) * CC + c0];
        int kc = k >> 5, kk = k & 31, idx = (kk & 7) | ((kk >> 1) & 8), lb = (kk & 8) << 1;
#pragma unroll
        for (int i = 0; i < 4; ++i) {
            int c = c0 + i;
            *(unsigned int*)&wl[kc][c >> 4][lb + (c & 15)][idx] = packbf2(wlo[i], whi[i]);
        }
    }

    // staging pipeline registers (bf16-pair packed x chunk slice for this thread)
    unsigned int xstage[16];
    auto load_x = [&](int k0) {          // global loads + convert, no LDS traffic
#pragma unroll
        for (int g = 0; g < 4; ++g) {
            int G  = g * 128 + t;        // 0..511 pair-groups
            int k  = (G >> 5) * 2;
            int c0 = (G & 31) * 4;
            const float* xp = x + (size_t)(b * NN + k0 + k) * CC + c0;
            f32x4 xlo = *(const f32x4*)xp;
            f32x4 xhi = *(const f32x4*)(xp + CC);
#pragma unroll
            for (int i = 0; i < 4; ++i) xstage[g * 4 + i] = packbf2(xlo[i], xhi[i]);
        }
    };
    auto store_x = [&](int buf) {        // scatter into B-operand layout
#pragma unroll
        for (int g = 0; g < 4; ++g) {
            int G  = g * 128 + t;
            int k  = (G >> 5) * 2;
            int c0 = (G & 31) * 4;
            int idx = (k & 7) | ((k >> 1) & 8), lb = (k & 8) << 1;
#pragma unroll
            for (int i = 0; i < 4; ++i) {
                int c = c0 + i;
                *(unsigned int*)&xb[buf][c >> 4][lb + (c & 15)][idx] = xstage[g * 4 + i];
            }
        }
    };

    load_x(0);
    store_x(0);
    __syncthreads();

    const float sjh = 0.5f * sjl[wave * 16 + n];

    v8f acc[8];
    v8f accS = {0.f, 0.f, 0.f, 0.f, 0.f, 0.f, 0.f, 0.f};
#pragma unroll
    for (int i = 0; i < 8; ++i) acc[i] = accS;

    V16U ones;
#pragma unroll
    for (int i = 0; i < 16; ++i) ones.s[i] = 0x3F80;     // bf16 1.0

    const int adjRowBase = (b * NN + j0 + wave * 16 + n) * NN;

    for (int kc = 0; kc < NCHUNK; ++kc) {
        const int k0  = kc * 32;
        const int cur = kc & 1;

        // issue next chunk's global loads early (overlaps with tanh/WMMA below)
        if (kc + 1 < NCHUNK) load_x(k0 + 32);

        // ---- A operand: masked sigmoid weights, streamed adj (non-temporal) ----
        const int  kA   = k0 + hi * 8;
        const int* arow = adj + adjRowBase + kA;
        i32x4 aa0 = __builtin_nontemporal_load((const i32x4*)(arow + 0));
        i32x4 aa1 = __builtin_nontemporal_load((const i32x4*)(arow + 4));
        i32x4 aa2 = __builtin_nontemporal_load((const i32x4*)(arow + 16));
        i32x4 aa3 = __builtin_nontemporal_load((const i32x4*)(arow + 20));
        __builtin_prefetch((const void*)(arow + 32), 0, 1);
        f32x4 g0 = *(const f32x4*)&skl[kA + 0]  + sjh;   // v_pk_add_f32
        f32x4 g1 = *(const f32x4*)&skl[kA + 4]  + sjh;
        f32x4 g2 = *(const f32x4*)&skl[kA + 16] + sjh;
        f32x4 g3 = *(const f32x4*)&skl[kA + 20] + sjh;

        float sc[16];
#pragma unroll
        for (int e = 0; e < 4; ++e) {
            sc[e]      = sigmoid_from_half_arg(g0[e]);
            sc[4 + e]  = sigmoid_from_half_arg(g1[e]);
            sc[8 + e]  = sigmoid_from_half_arg(g2[e]);
            sc[12 + e] = sigmoid_from_half_arg(g3[e]);
        }
        V16U A;
        A.u[0] = masked_pair(sc[0],  sc[1],  aa0[0], aa0[1]);
        A.u[1] = masked_pair(sc[2],  sc[3],  aa0[2], aa0[3]);
        A.u[2] = masked_pair(sc[4],  sc[5],  aa1[0], aa1[1]);
        A.u[3] = masked_pair(sc[6],  sc[7],  aa1[2], aa1[3]);
        A.u[4] = masked_pair(sc[8],  sc[9],  aa2[0], aa2[1]);
        A.u[5] = masked_pair(sc[10], sc[11], aa2[2], aa2[3]);
        A.u[6] = masked_pair(sc[12], sc[13], aa3[0], aa3[1]);
        A.u[7] = masked_pair(sc[14], sc[15], aa3[2], aa3[3]);

        // ---- 8 support WMMAs + 1 rowsum WMMA ----
#pragma unroll
        for (int ct = 0; ct < 8; ++ct) {
            v16bf bm = *(const v16bf*)(&xb[cur][ct][lane][0]);
            acc[ct] = __builtin_amdgcn_wmma_f32_16x16x32_bf16(
                false, A.v, false, bm, (short)0, acc[ct], false, false);
        }
        accS = __builtin_amdgcn_wmma_f32_16x16x32_bf16(
            false, A.v, false, ones.v, (short)0, accS, false, false);

        // write next chunk into the other buffer AFTER this chunk's B loads
        if (kc + 1 < NCHUNK) store_x(cur ^ 1);
        __syncthreads();
    }

    // ---- normalize (rowsum sits in matching D-layout slot) & stage support ----
    float inv[8];
#pragma unroll
    for (int r = 0; r < 8; ++r) inv[r] = __builtin_amdgcn_rcpf(accS[r]);
#pragma unroll
    for (int ct = 0; ct < 8; ++ct)
#pragma unroll
        for (int r = 0; r < 8; ++r) {
            int Mr = r + hi * 8;
            supp[wave][Mr][ct * 16 + n] = (__bf16)(acc[ct][r] * inv[r]);   // v_fma_mix*_bf16
        }
    __syncthreads();

    // ---- second GEMM: out_tile = support(16x128) @ W_lin + b_lin ----
#pragma unroll
    for (int ct = 0; ct < 8; ++ct) {
        float bl = blin[ct * 16 + n];
        v8f c2 = {bl, bl, bl, bl, bl, bl, bl, bl};
#pragma unroll
        for (int kc2 = 0; kc2 < 4; ++kc2) {
            int base = kc2 * 32 + hi * 8;
            V16U Av;
            Av.q[0] = *(const u32x4*)&supp[wave][n][base];
            Av.q[1] = *(const u32x4*)&supp[wave][n][base + 16];
            v16bf wb = *(const v16bf*)(&wl[kc2][ct][lane][0]);
            c2 = __builtin_amdgcn_wmma_f32_16x16x32_bf16(
                false, Av.v, false, wb, (short)0, c2, false, false);
        }
#pragma unroll
        for (int r = 0; r < 8; ++r) {
            int Mr = r + hi * 8;
            out[(size_t)(b * NN + j0 + wave * 16 + Mr) * CC + ct * 16 + n] = c2[r];
        }
    }
}

// ---------------------------------------------------------------------------
extern "C" void kernel_launch(void* const* d_in, const int* in_sizes, int n_in,
                              void* d_out, int out_size, void* d_ws, size_t ws_size,
                              hipStream_t stream) {
    const float* x     = (const float*)d_in[0];
    const int*   adj   = (const int*)  d_in[1];
    const float* W_fc  = (const float*)d_in[2];
    const float* b_fc  = (const float*)d_in[3];
    const float* W_lin = (const float*)d_in[4];
    const float* b_lin = (const float*)d_in[5];
    float*       out   = (float*)d_out;

    float* sj = (float*)d_ws;                 // B*N floats
    float* sk = sj + BATCH * NN;              // B*N floats

    gc_scores<<<(BATCH * NN) / 256, 256, 0, stream>>>(x, W_fc, sj, sk);

    dim3 grid(NN / 64, BATCH);
    gc_fused<<<grid, 128, 0, stream>>>(x, adj, sj, sk, b_fc, W_lin, b_lin, out);
}
